// DiffusionLoss_7937099563259
// MI455X (gfx1250) — matrix operand order repaired and checked by reading, not compiled, not run on hardware
//
#include <hip/hip_runtime.h>
#include <hip/hip_bf16.h>

typedef __attribute__((ext_vector_type(16))) __bf16 v16bf;
typedef __attribute__((ext_vector_type(8)))  float  v8f;

#define TILE 128
#define BK 32
#define LDSS 40   // LDS row stride in bf16 elements (80B, 16B-aligned rows)

// ---------- fragment extraction from LDS (row-major [row][k] tiles) ----------
// A-matrix 16x32 bf16 layout (ISA 7.12.2): lane m=L%16, half=L/16;
// element e: k = e + 8*half for e<8, k = 16 + (e-8) + 8*half for e>=8.
__device__ __forceinline__ v16bf frag_a(const __bf16* rowp, int half) {
    v16bf f;
#pragma unroll
    for (int i = 0; i < 8; ++i) {
        f[i]     = rowp[8 * half + i];
        f[8 + i] = rowp[16 + 8 * half + i];
    }
    return f;
}
// B-matrix 32x16 bf16 layout: lane n=L%16, half=L/16; element e: k = e + 8*half for e<8, k = 16 + (e-8) + 8*half for e>=8.
__device__ __forceinline__ v16bf frag_b(const __bf16* rowp, int half) {
    v16bf f;
#pragma unroll
    for (int i = 0; i < 16; ++i) f[i] = rowp[(i < 8 ? 0 : 8) + 8 * half + i];
    return f;
}

// ---------- async global -> LDS: 32 contiguous bytes per lane ----------
// GVS mode: per-lane LDS address VGPR, per-lane 32-bit byte offset VGPR,
// uniform 64-bit base in SGPR pair. inst offset applies to BOTH sides
// (LDS and global), so offset:16 fetches the second 16B of the same slab.
__device__ __forceinline__ void async_tile32(unsigned lds, unsigned goff,
                                             const __bf16* gbase) {
    asm volatile(
        "global_load_async_to_lds_b128 %0, %1, %2 offset:0\n\t"
        "global_load_async_to_lds_b128 %0, %1, %2 offset:16"
        :: "v"(lds), "v"(goff), "s"(gbase) : "memory");
}
__device__ __forceinline__ unsigned lds_addr(const void* p) {
    // generic pointer to LDS = {shared aperture, 32-bit LDS byte offset}
    return (unsigned)(uintptr_t)p;
}

// ---------- split-bf16 GEMM: (Chi+Clo) = A * B (+ diagAdd on diagonal) ----------
// A = Ahi + Alo, B = Bhi + Blo (bf16 hi/lo split). B is SYMMETRIC, so its rows
// are its columns: B tiles use the same contiguous row-major path as A tiles.
// Output is written directly as a bf16 hi/lo split (fused split epilogue).
// Double-buffered LDS + async loads: chunk c+1 streams in while chunk c computes.
__global__ __launch_bounds__(256)
void gemm_bf16x3(const __bf16* __restrict__ Ahi, const __bf16* __restrict__ Alo,
                 const __bf16* __restrict__ Bhi, const __bf16* __restrict__ Blo,
                 __bf16* __restrict__ Chi, __bf16* __restrict__ Clo,
                 int n, float diagAdd)
{
    __shared__ __align__(16) __bf16 lAhi[2][TILE][LDSS];
    __shared__ __align__(16) __bf16 lAlo[2][TILE][LDSS];
    __shared__ __align__(16) __bf16 lBhi[2][TILE][LDSS];
    __shared__ __align__(16) __bf16 lBlo[2][TILE][LDSS];

    const int tid  = threadIdx.x;
    const int lane = tid & 31;
    const int wave = tid >> 5;
    const int wr   = wave >> 1;   // 0..3 : wave's 32-row strip
    const int wc   = wave & 1;    // 0..1 : wave's 64-col strip
    const int lr   = lane & 15;
    const int lh   = lane >> 4;

    const int rowBase = blockIdx.y * TILE;
    const int colBase = blockIdx.x * TILE;

    const int ldr = tid >> 1;     // 0..127 : row loaded by this thread
    const int ldh = tid & 1;      // which 16-element half of the 32-wide k-slab

    // per-thread global byte offsets (bf16 = 2B), advanced by 2*BK per chunk
    const unsigned rowOffA = (unsigned)(((rowBase + ldr) * n + 16 * ldh) * 2);
    const unsigned rowOffB = (unsigned)(((colBase + ldr) * n + 16 * ldh) * 2); // symmetric trick

    auto issue = [&](int k0, int buf) {
        const unsigned ga = rowOffA + (unsigned)(2 * k0);
        const unsigned gb = rowOffB + (unsigned)(2 * k0);
        async_tile32(lds_addr(&lAhi[buf][ldr][16 * ldh]), ga, Ahi);
        async_tile32(lds_addr(&lAlo[buf][ldr][16 * ldh]), ga, Alo);
        async_tile32(lds_addr(&lBhi[buf][ldr][16 * ldh]), gb, Bhi);
        async_tile32(lds_addr(&lBlo[buf][ldr][16 * ldh]), gb, Blo);
    };

    v8f acc[2][4] = {};
    const int nChunks = n / BK;

    issue(0, 0);                                   // prologue: chunk 0 -> buf 0

    for (int c = 0; c < nChunks; ++c) {
        const int cur = c & 1;
        if (c + 1 < nChunks) {
            issue((c + 1) * BK, cur ^ 1);          // prefetch next chunk
            asm volatile("s_wait_asynccnt 0x8" ::: "memory");  // chunk c landed
        } else {
            asm volatile("s_wait_asynccnt 0x0" ::: "memory");
        }
        __syncthreads();                           // all waves' data visible

        v16bf fAhi[2], fAlo[2];
#pragma unroll
        for (int mt = 0; mt < 2; ++mt) {
            const int r = wr * 32 + mt * 16 + lr;
            fAhi[mt] = frag_a(&lAhi[cur][r][0], lh);
            fAlo[mt] = frag_a(&lAlo[cur][r][0], lh);
        }
#pragma unroll
        for (int nt = 0; nt < 4; ++nt) {
            const int r = wc * 64 + nt * 16 + lr;
            v16bf fBhi = frag_b(&lBhi[cur][r][0], lh);
            v16bf fBlo = frag_b(&lBlo[cur][r][0], lh);
#pragma unroll
            for (int mt = 0; mt < 2; ++mt) {
                acc[mt][nt] = __builtin_amdgcn_wmma_f32_16x16x32_bf16(
                    false, fAhi[mt], false, fBhi, (short)0, acc[mt][nt], false, false);
                acc[mt][nt] = __builtin_amdgcn_wmma_f32_16x16x32_bf16(
                    false, fAhi[mt], false, fBlo, (short)0, acc[mt][nt], false, false);
                acc[mt][nt] = __builtin_amdgcn_wmma_f32_16x16x32_bf16(
                    false, fAlo[mt], false, fBhi, (short)0, acc[mt][nt], false, false);
            }
        }
        __syncthreads();   // buf cur free before chunk c+2 streams into it
    }

    // Epilogue: C/D 16x16 f32 layout — col = lane%16, row = vgpr + 8*(lane/16).
    // Apply diagonal add, then write the bf16 hi/lo split directly.
#pragma unroll
    for (int mt = 0; mt < 2; ++mt) {
#pragma unroll
        for (int nt = 0; nt < 4; ++nt) {
            const int col = colBase + wc * 64 + nt * 16 + lr;
#pragma unroll
            for (int r = 0; r < 8; ++r) {
                const int row = rowBase + wr * 32 + mt * 16 + r + 8 * lh;
                float v = acc[mt][nt][r];
                if (row == col) v += diagAdd;
                const __bf16 h = (__bf16)v;
                const size_t idx = (size_t)row * n + col;
                Chi[idx] = h;
                Clo[idx] = (__bf16)(v - (float)h);
            }
        }
    }
}

// ---------- degree: deg[i] = sum_{j!=i} sigmoid((50-d_ij)/50) ----------
__global__ __launch_bounds__(256)
void deg_kernel(const float* __restrict__ pos, float* __restrict__ deg, int n)
{
    __shared__ float red[256];
    const int i = blockIdx.x;
    const float xi = pos[3 * i], yi = pos[3 * i + 1], zi = pos[3 * i + 2];
    float s = 0.f;
    for (int j = threadIdx.x; j < n; j += 256) {
        if (j == i) continue;
        const float dx = xi - pos[3 * j];
        const float dy = yi - pos[3 * j + 1];
        const float dz = zi - pos[3 * j + 2];
        const float d = sqrtf(dx * dx + dy * dy + dz * dz);
        s += 1.f / (1.f + expf((d - 50.f) / 50.f));
    }
    red[threadIdx.x] = s;
    __syncthreads();
    for (int w = 128; w > 0; w >>= 1) {
        if (threadIdx.x < w) red[threadIdx.x] += red[threadIdx.x + w];
        __syncthreads();
    }
    if (threadIdx.x == 0) deg[i] = red[0];
}

// ---------- A = (-tau/2^s)*norm_lap, written directly as bf16 hi/lo split ----------
__global__ __launch_bounds__(256)
void build_A_split(const float* __restrict__ pos, const float* __restrict__ deg,
                   __bf16* __restrict__ hi, __bf16* __restrict__ lo, int n, float scale)
{
    const int bpr = n / 256;
    const int i = blockIdx.x / bpr;
    const int j = (blockIdx.x % bpr) * 256 + threadIdx.x;
    float lap;
    if (i == j) {
        lap = deg[i];
    } else {
        const float dx = pos[3 * i]     - pos[3 * j];
        const float dy = pos[3 * i + 1] - pos[3 * j + 1];
        const float dz = pos[3 * i + 2] - pos[3 * j + 2];
        const float d = sqrtf(dx * dx + dy * dy + dz * dz);
        lap = -1.f / (1.f + expf((d - 50.f) / 50.f));
    }
    const float v = scale * lap * rsqrtf(deg[i] + 1e-6f) * rsqrtf(deg[j] + 1e-6f);
    const __bf16 h = (__bf16)v;
    const size_t idx = (size_t)i * n + j;
    hi[idx] = h;
    lo[idx] = (__bf16)(v - (float)h);
}

// ---------- Horner init: P = c_m * A + c_{m-1} * I, emitted as split ----------
__global__ __launch_bounds__(256)
void horner_init(const __bf16* __restrict__ Ahi, const __bf16* __restrict__ Alo,
                 __bf16* __restrict__ Phi, __bf16* __restrict__ Plo,
                 int n, float cm, float cm1)
{
    const int bpr = n / 256;
    const int i = blockIdx.x / bpr;
    const int j = (blockIdx.x % bpr) * 256 + threadIdx.x;
    const size_t idx = (size_t)i * n + j;
    float v = cm * ((float)Ahi[idx] + (float)Alo[idx]);
    if (i == j) v += cm1;
    const __bf16 h = (__bf16)v;
    Phi[idx] = h;
    Plo[idx] = (__bf16)(v - (float)h);
}

// ---------- column stats on a split matrix (H = hi + lo) ----------
__global__ __launch_bounds__(256)
void col_mean(const __bf16* __restrict__ hi, const __bf16* __restrict__ lo,
              float* __restrict__ mean, int n)
{
    const int j = blockIdx.x * 256 + threadIdx.x;
    float s = 0.f;
    for (int i = 0; i < n; ++i) {
        const size_t idx = (size_t)i * n + j;
        s += (float)hi[idx] + (float)lo[idx];
    }
    mean[j] = s / (float)n;
}

__global__ __launch_bounds__(256)
void col_cv(const __bf16* __restrict__ hi, const __bf16* __restrict__ lo,
            const float* __restrict__ mean, float* __restrict__ cv, int n)
{
    const int j = blockIdx.x * 256 + threadIdx.x;
    const float m = mean[j];
    float s = 0.f;
    for (int i = 0; i < n; ++i) {
        const size_t idx = (size_t)i * n + j;
        const float d = ((float)hi[idx] + (float)lo[idx]) - m;
        s += d * d;
    }
    cv[j] = sqrtf(s / (float)(n - 1)) / (m + 1e-6f);   // ddof=1, matches torch.std
}

__global__ __launch_bounds__(256)
void reduce_add(const float* __restrict__ cv, float* __restrict__ accum, int n)
{
    __shared__ float red[256];
    float s = 0.f;
    for (int j = threadIdx.x; j < n; j += 256) s += cv[j];
    red[threadIdx.x] = s;
    __syncthreads();
    for (int w = 128; w > 0; w >>= 1) {
        if (threadIdx.x < w) red[threadIdx.x] += red[threadIdx.x + w];
        __syncthreads();
    }
    if (threadIdx.x == 0) accum[0] += red[0];   // single block -> deterministic
}

__global__ void zero_kernel(float* __restrict__ a) { a[0] = 0.f; }

__global__ void finalize_kernel(const float* __restrict__ accum, float* __restrict__ out, int n)
{
    out[0] = accum[0] / (2.f * (float)n);
}

// =============================== launch ===============================
extern "C" void kernel_launch(void* const* d_in, const int* in_sizes, int n_in,
                              void* d_out, int out_size, void* d_ws, size_t ws_size,
                              hipStream_t stream)
{
    const int n = in_sizes[0] / 3;                 // 4096
    const float* pos = (const float*)d_in[0];
    float* out = (float*)d_out;

    char* ws = (char*)d_ws;
    const size_t szB = (size_t)n * n * sizeof(__bf16);
    __bf16* Ahi = (__bf16*)(ws);
    __bf16* Alo = (__bf16*)(ws + szB);
    __bf16* Phi = (__bf16*)(ws + 2 * szB);
    __bf16* Plo = (__bf16*)(ws + 3 * szB);
    __bf16* Qhi = (__bf16*)(ws + 4 * szB);
    __bf16* Qlo = (__bf16*)(ws + 5 * szB);
    float*  deg   = (float*)(ws + 6 * szB);
    float*  mean  = deg + n;
    float*  cv    = mean + n;
    float*  accum = cv + n;

    const int  elemBlocks = (int)(((size_t)n * n) / 256);
    const dim3 gGemm(n / TILE, n / TILE);
    const int  colBlocks = n / 256;

    // 1/k! for Taylor of exp, degree 8; scaling s=5 -> ||A/32|| <= 0.3125
    const float c[9] = {1.f, 1.f, 0.5f, 1.f / 6.f, 1.f / 24.f, 1.f / 120.f,
                        1.f / 720.f, 1.f / 5040.f, 1.f / 40320.f};

    zero_kernel<<<1, 1, 0, stream>>>(accum);
    deg_kernel<<<n, 256, 0, stream>>>(pos, deg, n);
    build_A_split<<<elemBlocks, 256, 0, stream>>>(pos, deg, Ahi, Alo, n, -5.f / 32.f);

    // Horner: P = (((c8*A + c7 I)A + c6 I)A + ... + c0 I)  — 7 GEMMs
    horner_init<<<elemBlocks, 256, 0, stream>>>(Ahi, Alo, Phi, Plo, n, c[8], c[7]);
    for (int k = 6; k >= 0; --k) {
        gemm_bf16x3<<<gGemm, 256, 0, stream>>>(Ahi, Alo, Phi, Plo, Qhi, Qlo, n, c[k]);
        __bf16* t;
        t = Phi; Phi = Qhi; Qhi = t;
        t = Plo; Plo = Qlo; Qlo = t;
    }
    // Squaring: P^(2^5) = expm(-5 * norm_lap)  — 5 GEMMs
    for (int sq = 0; sq < 5; ++sq) {
        gemm_bf16x3<<<gGemm, 256, 0, stream>>>(Phi, Plo, Phi, Plo, Qhi, Qlo, n, 0.f);
        __bf16* t;
        t = Phi; Phi = Qhi; Qhi = t;
        t = Plo; Plo = Qlo; Qlo = t;
    }

    // tau = 5 statistics
    col_mean<<<colBlocks, 256, 0, stream>>>(Phi, Plo, mean, n);
    col_cv<<<colBlocks, 256, 0, stream>>>(Phi, Plo, mean, cv, n);
    reduce_add<<<1, 256, 0, stream>>>(cv, accum, n);

    // tau = 10: heat10 = heat5 @ heat5  — 1 GEMM
    gemm_bf16x3<<<gGemm, 256, 0, stream>>>(Phi, Plo, Phi, Plo, Qhi, Qlo, n, 0.f);
    col_mean<<<colBlocks, 256, 0, stream>>>(Qhi, Qlo, mean, n);
    col_cv<<<colBlocks, 256, 0, stream>>>(Qhi, Qlo, mean, cv, n);
    reduce_add<<<1, 256, 0, stream>>>(cv, accum, n);

    finalize_kernel<<<1, 1, 0, stream>>>(accum, out, n);
}
